// AutoregressiveDecoder_12223476924789
// MI455X (gfx1250) — compile-verified, matrix operand order checked by reading, not run on hardware
//
#include <hip/hip_runtime.h>
#include <hip/hip_fp16.h>

typedef __attribute__((ext_vector_type(16))) _Float16 v16h;
typedef __attribute__((ext_vector_type(8)))  _Float16 v8h;
typedef __attribute__((ext_vector_type(8)))  float    v8f;

// ---------------- WMMA fragment helpers (gfx1250 16x16x32 f16, wave32) ----------
// A-matrix 16x32 (MxK), per ISA 7.12.2: lane L(0-15): row M=L, elems 0..7 = K+0..7,
// elems 8..15 = K+16..23 ; lane L(16-31): row M=L-16, K offset by +8.
__device__ __forceinline__ v16h frag_from_f32(const float* __restrict__ p, int koff) {
  v16h a;
#pragma unroll
  for (int i = 0; i < 8; ++i) {
    a[i]     = (_Float16)p[koff + i];
    a[i + 8] = (_Float16)p[koff + 16 + i];
  }
  return a;
}
__device__ __forceinline__ v16h frag_from_f16(const _Float16* __restrict__ p, int koff) {
  v16h b;
#pragma unroll
  for (int i = 0; i < 8; ++i) {
    b[i]     = p[koff + i];
    b[i + 8] = p[koff + 16 + i];
  }
  return b;
}

// ---------------- weight transpose + f16 cast:  src[K,N] f32 -> dst[N,K] f16 ----
__global__ void wcast_kernel(const float* __restrict__ src, _Float16* __restrict__ dst,
                             int K, int N) {
  int idx = blockIdx.x * blockDim.x + threadIdx.x;
  if (idx >= K * N) return;
  int n = idx / K, k = idx - n * K;
  dst[idx] = (_Float16)src[(size_t)k * N + n];
}

// ---------------- encoder K/V projection: [16384,256] x [256,1024] (x4 tensors) --
// grid (16, 1024, 4), block 128 (4 waves); one wave -> one 16x16 tile.
__global__ void enc_kv_kernel(const float* __restrict__ enc,
                              const _Float16* wt0, const float* b0, _Float16* o0,
                              const _Float16* wt1, const float* b1, _Float16* o1,
                              const _Float16* wt2, const float* b2, _Float16* o2,
                              const _Float16* wt3, const float* b3, _Float16* o3) {
  int z = blockIdx.z;
  const _Float16* WT = (z == 0) ? wt0 : (z == 1) ? wt1 : (z == 2) ? wt2 : wt3;
  const float*  bias = (z == 0) ? b0  : (z == 1) ? b1  : (z == 2) ? b2  : b3;
  _Float16*        O = (z == 0) ? o0  : (z == 1) ? o1  : (z == 2) ? o2  : o3;

  int lane  = threadIdx.x & 31;
  int ntile = blockIdx.x * (blockDim.x >> 5) + (threadIdx.x >> 5);  // 0..63
  int mtile = blockIdx.y;                                          // 0..1023
  int row = lane & 15, koff = (lane >> 4) * 8;
  const float*    arow = enc + (size_t)(mtile * 16 + row) * 256;
  const _Float16* brow = WT  + (size_t)(ntile * 16 + row) * 256;
  v8f c = {};
#pragma unroll
  for (int k0 = 0; k0 < 256; k0 += 32) {
    v16h a = frag_from_f32(arow + k0, koff);
    v16h b = frag_from_f16(brow + k0, koff);
    c = __builtin_amdgcn_wmma_f32_16x16x32_f16(false, a, false, b, (short)0, c, false, false);
  }
  int n = ntile * 16 + (lane & 15);
  int mbase = (lane >> 4) * 8;
  float bi = bias[n];
#pragma unroll
  for (int r = 0; r < 8; ++r) {
    int m = mtile * 16 + mbase + r;
    O[(size_t)m * 1024 + n] = (_Float16)(c[r] + bi);
  }
}

// ---------------- decode-side GEMM: Y[16,N] = A[16,K](f32) x WT[N,K](f16) --------
// grid.x * waves = N/16 ; block 128. bias/resid optional, relu optional.
__global__ void gemm16_kernel(const float* __restrict__ A, int lda,
                              const _Float16* __restrict__ WT,
                              const float* __restrict__ bias,
                              const float* resid, int ldr,
                              float* Y, int ldy, int K, int relu) {
  int lane  = threadIdx.x & 31;
  int ntile = blockIdx.x * (blockDim.x >> 5) + (threadIdx.x >> 5);
  int row = lane & 15, koff = (lane >> 4) * 8;
  const float*    arow = A  + (size_t)row * lda;
  const _Float16* brow = WT + (size_t)(ntile * 16 + row) * K;
  v8f c = {};
  for (int k0 = 0; k0 < K; k0 += 32) {
    v16h a = frag_from_f32(arow + k0, koff);
    v16h b = frag_from_f16(brow + k0, koff);
    c = __builtin_amdgcn_wmma_f32_16x16x32_f16(false, a, false, b, (short)0, c, false, false);
  }
  int n = ntile * 16 + (lane & 15);
  int mbase = (lane >> 4) * 8;
  float bi = bias ? bias[n] : 0.0f;
#pragma unroll
  for (int r = 0; r < 8; ++r) {
    int m = mbase + r;
    float o = c[r] + bi;
    if (resid) o += resid[(size_t)m * ldr + n];
    if (relu)  o = fmaxf(o, 0.0f);
    Y[(size_t)m * ldy + n] = o;
  }
}

// ---------------- self attention over cached K/V (T <= 10) -----------------------
__global__ void self_attn_kernel(const float* __restrict__ q, const float* __restrict__ kc,
                                 const float* __restrict__ vc, float* __restrict__ ctx, int T) {
  int b = blockIdx.x, h = blockIdx.y, tid = threadIdx.x;
  __shared__ float sl[16];
  const float* qv = q + b * 1024 + h * 256;
  if (tid < T) {
    const float* kr = kc + ((size_t)b * 10 + tid) * 1024 + h * 256;
    float acc = 0.0f;
    for (int d = 0; d < 256; ++d) acc += qv[d] * kr[d];
    sl[tid] = acc * 0.0625f;  // 1/sqrt(256)
  }
  __syncthreads();
  float m = -1e30f;
  for (int j = 0; j < T; ++j) m = fmaxf(m, sl[j]);
  float w[10], sum = 0.0f;
  for (int j = 0; j < T; ++j) { w[j] = __expf(sl[j] - m); sum += w[j]; }
  float inv = 1.0f / sum;
  float acc = 0.0f;
  for (int j = 0; j < T; ++j)
    acc += w[j] * vc[((size_t)b * 10 + j) * 1024 + h * 256 + tid];
  ctx[b * 1024 + h * 256 + tid] = acc * inv;
}

// ---------------- cross attention vs L2-resident f16 enc K/V (S=1024) ------------
__global__ void cross_attn_kernel(const float* __restrict__ q, const _Float16* __restrict__ Kh,
                                  const _Float16* __restrict__ Vh, float* __restrict__ ctx) {
  int b = blockIdx.x, h = blockIdx.y, tid = threadIdx.x;
  __shared__ float qs[256];
  __shared__ float wl[1024];
  __shared__ float red[256];
  qs[tid] = q[b * 1024 + h * 256 + tid];
  __syncthreads();
  const _Float16* Kb = Kh + (size_t)b * 1024 * 1024 + h * 256;
#pragma unroll
  for (int i = 0; i < 4; ++i) {
    int s = i * 256 + tid;
    const v8h* kr8 = (const v8h*)(Kb + (size_t)s * 1024);
    float acc = 0.0f;
    for (int d8 = 0; d8 < 32; ++d8) {
      v8h kv = kr8[d8];
#pragma unroll
      for (int j = 0; j < 8; ++j) acc += qs[d8 * 8 + j] * (float)kv[j];
    }
    wl[s] = acc * 0.0625f;
  }
  __syncthreads();
  float m = fmaxf(fmaxf(wl[tid], wl[tid + 256]), fmaxf(wl[tid + 512], wl[tid + 768]));
  red[tid] = m; __syncthreads();
  for (int s = 128; s > 0; s >>= 1) { if (tid < s) red[tid] = fmaxf(red[tid], red[tid + s]); __syncthreads(); }
  m = red[0];
  __syncthreads();
  float lsum = 0.0f;
#pragma unroll
  for (int i = 0; i < 4; ++i) {
    int s = i * 256 + tid;
    float e = __expf(wl[s] - m);
    wl[s] = e; lsum += e;
  }
  red[tid] = lsum; __syncthreads();
  for (int s = 128; s > 0; s >>= 1) { if (tid < s) red[tid] += red[tid + s]; __syncthreads(); }
  float inv = 1.0f / red[0];
  const _Float16* Vb = Vh + (size_t)b * 1024 * 1024 + h * 256;
  float acc = 0.0f;
  for (int s = 0; s < 1024; ++s) acc += wl[s] * (float)Vb[(size_t)s * 1024 + tid];
  ctx[b * 1024 + h * 256 + tid] = acc * inv;
}

// ---------------- LayerNorm over E=256 (eps 1e-3, population var) ----------------
__global__ void ln_kernel(const float* __restrict__ y, const float* __restrict__ g,
                          const float* __restrict__ bb, float* __restrict__ x) {
  __shared__ float rs[256], rq[256];
  int b = blockIdx.x, tid = threadIdx.x;
  float v = y[b * 256 + tid];
  rs[tid] = v; rq[tid] = v * v; __syncthreads();
  for (int s = 128; s > 0; s >>= 1) {
    if (tid < s) { rs[tid] += rs[tid + s]; rq[tid] += rq[tid + s]; }
    __syncthreads();
  }
  float mean = rs[0] * (1.0f / 256.0f);
  float var  = rq[0] * (1.0f / 256.0f) - mean * mean;
  float inv  = rsqrtf(var + 1e-3f);
  x[b * 256 + tid] = (v - mean) * inv * g[tid] + bb[tid];
}

// ---------------- output head + next-token embedding -----------------------------
__global__ void final_kernel(const float* x, const float* __restrict__ outW,
                             const float* __restrict__ outB, const float* __restrict__ coordW,
                             const float* __restrict__ coordB, const float* __restrict__ pos,
                             int t, float* __restrict__ out, float* xnext) {
  int b = blockIdx.x, tid = threadIdx.x;
  __shared__ float r0[256], r1[256];
  float v = x[b * 256 + tid];
  r0[tid] = v * outW[tid * 2 + 0];
  r1[tid] = v * outW[tid * 2 + 1];
  __syncthreads();
  for (int s = 128; s > 0; s >>= 1) {
    if (tid < s) { r0[tid] += r0[tid + s]; r1[tid] += r1[tid + s]; }
    __syncthreads();
  }
  float nc0 = r0[0] + outB[0];
  float nc1 = r1[0] + outB[1];
  if (tid == 0) { out[(b * 10 + t) * 2 + 0] = nc0; out[(b * 10 + t) * 2 + 1] = nc1; }
  xnext[b * 256 + tid] = nc0 * coordW[tid] + nc1 * coordW[256 + tid]
                       + coordB[tid] + pos[(t + 1) * 256 + tid];
}

__global__ void init_kernel(const float* __restrict__ coordB, const float* __restrict__ pos,
                            float* __restrict__ x) {
  int b = blockIdx.x, tid = threadIdx.x;
  x[b * 256 + tid] = coordB[tid] + pos[tid];  // zeros @ coord_W == 0; + pos[0]
}

// =================================================================================
extern "C" void kernel_launch(void* const* d_in, const int* in_sizes, int n_in,
                              void* d_out, int out_size, void* d_ws, size_t ws_size,
                              hipStream_t stream) {
  (void)in_sizes; (void)n_in; (void)out_size; (void)ws_size;
  // JAX pytree leaf order (dicts sorted by key):
  const float* enc     = (const float*)d_in[0];   // [16,1024,256]
  const float* coord_W = (const float*)d_in[1];   // [2,256]
  const float* coord_b = (const float*)d_in[2];   // [256]
  struct LayerP {
    const float *W1, *W2, *b1, *b2, *bf1, *bf2;
    const float *caWk, *caWo, *caWq, *caWv, *cabk, *cabo, *cabq, *cabv;
    const float *g1, *g2;
    const float *saWk, *saWo, *saWq, *saWv, *sabk, *sabo, *sabq, *sabv;
  } L[2];
  for (int l = 0; l < 2; ++l) {
    int i = 3 + l * 24;
    L[l].W1  = (const float*)d_in[i + 0];  L[l].W2  = (const float*)d_in[i + 1];
    L[l].b1  = (const float*)d_in[i + 2];  L[l].b2  = (const float*)d_in[i + 3];
    L[l].bf1 = (const float*)d_in[i + 4];  L[l].bf2 = (const float*)d_in[i + 5];
    L[l].caWk = (const float*)d_in[i + 6]; L[l].caWo = (const float*)d_in[i + 7];
    L[l].caWq = (const float*)d_in[i + 8]; L[l].caWv = (const float*)d_in[i + 9];
    L[l].cabk = (const float*)d_in[i + 10]; L[l].cabo = (const float*)d_in[i + 11];
    L[l].cabq = (const float*)d_in[i + 12]; L[l].cabv = (const float*)d_in[i + 13];
    L[l].g1  = (const float*)d_in[i + 14]; L[l].g2  = (const float*)d_in[i + 15];
    L[l].saWk = (const float*)d_in[i + 16]; L[l].saWo = (const float*)d_in[i + 17];
    L[l].saWq = (const float*)d_in[i + 18]; L[l].saWv = (const float*)d_in[i + 19];
    L[l].sabk = (const float*)d_in[i + 20]; L[l].sabo = (const float*)d_in[i + 21];
    L[l].sabq = (const float*)d_in[i + 22]; L[l].sabv = (const float*)d_in[i + 23];
  }
  const float* out_W = (const float*)d_in[51];  // [256,2]
  const float* out_b = (const float*)d_in[52];  // [2]
  const float* pos   = (const float*)d_in[53];  // [11,256]
  float* out = (float*)d_out;                   // [16,10,2]

  // ---- workspace carve-up (256B aligned chunks) ----
  char* ws = (char*)d_ws;
  size_t off = 0;
  auto alloc = [&](size_t bytes) -> char* {
    char* p = ws + off;
    off = (off + bytes + 255) & ~(size_t)255;
    return p;
  };
  const size_t WEL = 262144;  // 256*1024 elems per weight matrix
  // per layer: [0]saWq [1]saWk [2]saWv [3]saWoT [4]caWq [5]caWk [6]caWv [7]caWoT [8]W1 [9]W2T
  _Float16* wt[2][10];
  for (int l = 0; l < 2; ++l)
    for (int i = 0; i < 10; ++i) wt[l][i] = (_Float16*)alloc(WEL * 2);
  _Float16* encK[2]; _Float16* encV[2];
  for (int l = 0; l < 2; ++l) {
    encK[l] = (_Float16*)alloc((size_t)16 * 1024 * 1024 * 2);  // 33.5MB, f16 -> L2 resident
    encV[l] = (_Float16*)alloc((size_t)16 * 1024 * 1024 * 2);
  }
  float* skc[2]; float* svc[2];
  for (int l = 0; l < 2; ++l) {
    skc[l] = (float*)alloc((size_t)16 * 10 * 1024 * 4);
    svc[l] = (float*)alloc((size_t)16 * 10 * 1024 * 4);
  }
  float* cur_x   = (float*)alloc(16 * 256 * 4);
  float* ybuf    = (float*)alloc(16 * 256 * 4);
  float* qs_buf  = (float*)alloc(16 * 1024 * 4);
  float* qc_buf  = (float*)alloc(16 * 1024 * 4);
  float* ctx_buf = (float*)alloc(16 * 1024 * 4);
  float* hbuf    = (float*)alloc(16 * 1024 * 4);

  // ---- phase 1: transpose-cast all weights to f16 [N,K] ----
  struct WJ { const float* src; _Float16* dst; int K, N; };
  for (int l = 0; l < 2; ++l) {
    WJ jobs[10] = {
      {L[l].saWq, wt[l][0], 256, 1024}, {L[l].saWk, wt[l][1], 256, 1024},
      {L[l].saWv, wt[l][2], 256, 1024}, {L[l].saWo, wt[l][3], 1024, 256},
      {L[l].caWq, wt[l][4], 256, 1024}, {L[l].caWk, wt[l][5], 256, 1024},
      {L[l].caWv, wt[l][6], 256, 1024}, {L[l].caWo, wt[l][7], 1024, 256},
      {L[l].W1,   wt[l][8], 256, 1024}, {L[l].W2,   wt[l][9], 1024, 256},
    };
    for (int j = 0; j < 10; ++j)
      wcast_kernel<<<dim3((jobs[j].K * jobs[j].N + 255) / 256), dim3(256), 0, stream>>>(
          jobs[j].src, jobs[j].dst, jobs[j].K, jobs[j].N);
  }

  // ---- phase 2: encoder K/V (hoisted, bandwidth-bound WMMA GEMM, f16 out) ----
  enc_kv_kernel<<<dim3(16, 1024, 4), dim3(128), 0, stream>>>(
      enc,
      wt[0][5], L[0].cabk, encK[0],
      wt[0][6], L[0].cabv, encV[0],
      wt[1][5], L[1].cabk, encK[1],
      wt[1][6], L[1].cabv, encV[1]);

  // ---- phase 3: incremental autoregressive decode (mathematically == reference) ----
  init_kernel<<<dim3(16), dim3(256), 0, stream>>>(coord_b, pos, cur_x);
  for (int t = 0; t < 10; ++t) {
    int T = t + 1;
    for (int l = 0; l < 2; ++l) {
      // self-attn QKV projections (K,V appended to cache slot t)
      gemm16_kernel<<<dim3(16), dim3(128), 0, stream>>>(cur_x, 256, wt[l][0], L[l].sabq,
                                                        nullptr, 0, qs_buf, 1024, 256, 0);
      gemm16_kernel<<<dim3(16), dim3(128), 0, stream>>>(cur_x, 256, wt[l][1], L[l].sabk,
                                                        nullptr, 0, skc[l] + t * 1024, 10240, 256, 0);
      gemm16_kernel<<<dim3(16), dim3(128), 0, stream>>>(cur_x, 256, wt[l][2], L[l].sabv,
                                                        nullptr, 0, svc[l] + t * 1024, 10240, 256, 0);
      self_attn_kernel<<<dim3(16, 4), dim3(256), 0, stream>>>(qs_buf, skc[l], svc[l], ctx_buf, T);
      gemm16_kernel<<<dim3(4), dim3(128), 0, stream>>>(ctx_buf, 1024, wt[l][3], L[l].sabo,
                                                       cur_x, 256, ybuf, 256, 1024, 0);
      ln_kernel<<<dim3(16), dim3(256), 0, stream>>>(ybuf, L[l].g1, L[l].b1, cur_x);
      // cross-attn
      gemm16_kernel<<<dim3(16), dim3(128), 0, stream>>>(cur_x, 256, wt[l][4], L[l].cabq,
                                                        nullptr, 0, qc_buf, 1024, 256, 0);
      cross_attn_kernel<<<dim3(16, 4), dim3(256), 0, stream>>>(qc_buf, encK[l], encV[l], ctx_buf);
      gemm16_kernel<<<dim3(4), dim3(128), 0, stream>>>(ctx_buf, 1024, wt[l][7], L[l].cabo,
                                                       cur_x, 256, ybuf, 256, 1024, 0);
      ln_kernel<<<dim3(16), dim3(256), 0, stream>>>(ybuf, L[l].g2, L[l].b2, cur_x);
      // MLP
      gemm16_kernel<<<dim3(16), dim3(128), 0, stream>>>(cur_x, 256, wt[l][8], L[l].bf1,
                                                        nullptr, 0, hbuf, 1024, 256, 1);
      gemm16_kernel<<<dim3(4), dim3(128), 0, stream>>>(hbuf, 1024, wt[l][9], L[l].bf2,
                                                       cur_x, 256, cur_x, 256, 1024, 0);
    }
    final_kernel<<<dim3(16), dim3(256), 0, stream>>>(cur_x, out_W, out_b, coord_W, coord_b,
                                                     pos, t, out, cur_x);
  }
}
